// ResidualGCN_61048665145406
// MI455X (gfx1250) — matrix operand order repaired and checked by reading, not compiled, not run on hardware
//
#include <hip/hip_runtime.h>
#include <hip/hip_bf16.h>
#include <cstdint>
#include <cstddef>

typedef __bf16 bf16_t;
typedef __attribute__((ext_vector_type(16))) __bf16 v16bf;
typedef __attribute__((ext_vector_type(8)))  __bf16 v8bf;
typedef __attribute__((ext_vector_type(2)))  __bf16 bf2;
typedef __attribute__((ext_vector_type(8)))  float  v8f;

// Types matching the async-to-LDS builtin signature:
// param0: global (AS1) int4*, param1: LDS (AS3) int4*
typedef int v4i __attribute__((vector_size(16)));
typedef __attribute__((address_space(1))) v4i as1_v4i;
typedef __attribute__((address_space(3))) v4i as3_v4i;

#define NEG_SLOPE 0.01f

#if __has_builtin(__builtin_amdgcn_global_load_async_to_lds_b128) && \
    __has_builtin(__builtin_amdgcn_s_wait_asynccnt)
#define USE_ASYNC_LDS 1
#else
#define USE_ASYNC_LDS 0
#endif

__device__ __forceinline__ float lrelu_f(float v) { return v >= 0.f ? v : NEG_SLOPE * v; }

// ---------------------------------------------------------------------------
// Pack a row-major fp32 weight [Kin x Kout] into bf16 WMMA B-fragment order.
// Fragment (kc, nt) covers K rows [kc*32, kc*32+32) and N cols [nt*16, nt*16+16).
// Within a fragment: lane L, element e -> W[kc*32 + (L>=16?16:0) + e][nt*16 + (L&15)]
// ---------------------------------------------------------------------------
__global__ void pack_w_kernel(const float* __restrict__ W, bf16_t* __restrict__ out,
                              int Kin, int Kout, int total) {
  int tid = blockIdx.x * blockDim.x + threadIdx.x;
  if (tid >= total) return;
  int e      = tid & 15;
  int lane   = (tid >> 4) & 31;
  int frag   = tid >> 9;
  int ntiles = Kout >> 4;
  int nt     = frag % ntiles;
  int kc     = frag / ntiles;
  int col    = (nt << 4) + (lane & 15);
  int k      = (kc << 5) + ((lane >> 4) << 4) + e;
  float v    = (k < Kin) ? W[(size_t)k * Kout + col] : 0.f;
  out[tid]   = (bf16_t)v;
}

// Convert x [N x 140] fp32 -> zero-padded [N x 160] bf16
__global__ void cvt_x_kernel(const float* __restrict__ x, bf16_t* __restrict__ xb, int N) {
  long long t = (long long)blockIdx.x * blockDim.x + threadIdx.x;
  long long total = (long long)N * 160;
  if (t >= total) return;
  int c = (int)(t % 160);
  long long i = t / 160;
  float v = (c < 140) ? x[i * 140 + c] : 0.f;
  xb[t] = (bf16_t)v;
}

// ---------------------------------------------------------------------------
// WMMA bf16 GEMM: C[N x Kout] = act(A[N x lda(bf16)] @ W + bias)
// Block: 256 threads = 8 waves (4 in M, 2 in N). Tile 128 rows x 32 cols.
// A staged into LDS with GLOBAL_LOAD_ASYNC_TO_LDS_B128 (ASYNCcnt) when the
// toolchain exposes the builtin; B from pre-packed bf16 fragments.
// ---------------------------------------------------------------------------
template <int ACT, int OUTBF>
__global__ __launch_bounds__(256) void gemm_wmma_kernel(
    const bf16_t* __restrict__ A, int lda,
    const bf16_t* __restrict__ Wp, const float* __restrict__ bias,
    void* __restrict__ C, int ldc, int N, int Kpad, int Kout) {
  __shared__ bf16_t As[128 * 40];

  const int lane   = threadIdx.x & 31;
  const int wave   = threadIdx.x >> 5;
  const int waveM  = wave & 3;
  const int waveN  = wave >> 2;
  const int ntiles = Kout >> 4;
  const int nt     = blockIdx.y * 2 + waveN;
  const int colB   = nt << 4;
  const long long rowBase = (long long)blockIdx.x * 128;
  const int hi8    = (lane >= 16) ? 8 : 0;

  // Per-thread staging coordinates: 2 threads per row, 32 bytes each.
  const int srow  = threadIdx.x >> 1;
  const int shalf = threadIdx.x & 1;
  const long long sgrow = rowBase + srow;
  const bool srow_ok = (sgrow < N);
  const bf16_t* sgbase = A + sgrow * (long long)lda + shalf * 16;
  bf16_t* slbase = &As[srow * 40 + shalf * 16];

  v8f acc0, acc1;
#pragma unroll
  for (int j = 0; j < 8; ++j) { acc0[j] = 0.f; acc1[j] = 0.f; }

  const int nkc = Kpad >> 5;
  for (int kc = 0; kc < nkc; ++kc) {
    // --- prefetch next K-tile of A (global_prefetch_b8) ---
    if (kc + 1 < nkc && srow_ok) {
      __builtin_prefetch((const void*)(sgbase + (kc + 1) * 32), 0, 1);
    }

    // --- stage A tile 128x32 bf16 into LDS ---
#if USE_ASYNC_LDS
    if (srow_ok) {
      as1_v4i* gp = (as1_v4i*)(sgbase + kc * 32);
      as3_v4i* lp = (as3_v4i*)slbase;
      __builtin_amdgcn_global_load_async_to_lds_b128(gp, lp, 0, 0);
      __builtin_amdgcn_global_load_async_to_lds_b128(gp, lp, 16, 0);
    }
    __builtin_amdgcn_s_wait_asynccnt(0);
#else
    {
      uint4 lo = make_uint4(0u, 0u, 0u, 0u);
      uint4 hi = make_uint4(0u, 0u, 0u, 0u);
      if (srow_ok) {
        const uint4* sp = (const uint4*)(sgbase + kc * 32);
        lo = sp[0];
        hi = sp[1];
      }
      uint4* dp = (uint4*)slbase;
      dp[0] = lo;
      dp[1] = hi;
    }
#endif
    __syncthreads();

    // --- B fragment: 16 contiguous bf16 per lane from packed weights ---
    v16bf bfrag;
    {
      const v8bf* bp = (const v8bf*)(Wp + (((size_t)(kc * ntiles + nt)) * 32 + lane) * 16);
      v8bf b0 = bp[0];
      v8bf b1 = bp[1];
#pragma unroll
      for (int e = 0; e < 8; ++e) { bfrag[e] = b0[e]; bfrag[e + 8] = b1[e]; }
    }

    // --- 2 M-subtiles per wave ---
#pragma unroll
    for (int sub = 0; sub < 2; ++sub) {
      int r = waveM * 32 + sub * 16 + (lane & 15);
      v16bf afrag;
#pragma unroll
      for (int v = 0; v < 8; ++v) {
        int kl = (hi8 << 1) + (v << 1);
        bf2 p = *(const bf2*)&As[r * 40 + kl];
        afrag[2 * v]     = p.x;
        afrag[2 * v + 1] = p.y;
      }
      if (sub == 0)
        acc0 = __builtin_amdgcn_wmma_f32_16x16x32_bf16(false, afrag, false, bfrag,
                                                       (short)0, acc0, false, false);
      else
        acc1 = __builtin_amdgcn_wmma_f32_16x16x32_bf16(false, afrag, false, bfrag,
                                                       (short)0, acc1, false, false);
    }
    __syncthreads();
  }

  // --- epilogue: bias + optional LeakyReLU, write bf16 or f32 ---
  int col = colB + (lane & 15);
  float bv = bias ? bias[col] : 0.f;
#pragma unroll
  for (int sub = 0; sub < 2; ++sub) {
#pragma unroll
    for (int j = 0; j < 8; ++j) {
      float a = sub ? acc1[j] : acc0[j];
      long long row = rowBase + waveM * 32 + sub * 16 + ((lane >> 4) << 3) + j;
      if (row < N) {
        float val = a + bv;
        if (ACT) val = lrelu_f(val);
        if (OUTBF) ((bf16_t*)C)[row * (long long)ldc + col] = (bf16_t)val;
        else       ((float*)C)[row * (long long)ldc + col]  = val;
      }
    }
  }
}

// ---------------------------------------------------------------------------
// Graph kernels
// ---------------------------------------------------------------------------
__global__ void deg_init_kernel(float* __restrict__ deg, int N) {
  int i = blockIdx.x * blockDim.x + threadIdx.x;
  if (i < N) deg[i] = 1.0f;  // self-loop
}

__global__ void deg_acc_kernel(const long long* __restrict__ dst, float* __restrict__ deg,
                               long long E) {
  long long e = (long long)blockIdx.x * blockDim.x + threadIdx.x;
  if (e < E) atomicAdd(&deg[(int)dst[e]], 1.0f);
}

// agg[dst] += hw[src] * rsqrt(deg[src]) * rsqrt(deg[dst])   (float4 per thread)
__global__ void scatter_kernel(const float* __restrict__ hw, const long long* __restrict__ src,
                               const long long* __restrict__ dst, const float* __restrict__ deg,
                               float* __restrict__ agg, long long E, int H) {
  long long tid = (long long)blockIdx.x * blockDim.x + threadIdx.x;
  int chunks = H >> 2;
  long long total = E * chunks;
  if (tid >= total) return;
  long long e = tid / chunks;
  int c = (int)(tid % chunks) << 2;
  int s = (int)src[e];
  int d = (int)dst[e];
  float nrm = rsqrtf(deg[s]) * rsqrtf(deg[d]);
  const float4 v = *(const float4*)(hw + (size_t)s * H + c);
  float* ap = agg + (size_t)d * H + c;
  atomicAdd(ap + 0, v.x * nrm);
  atomicAdd(ap + 1, v.y * nrm);
  atomicAdd(ap + 2, v.z * nrm);
  atomicAdd(ap + 3, v.w * nrm);
}

// h_out = lrelu(agg + hw/deg + bias) + ident     (bf16 out)
__global__ void finalize_kernel(const float* __restrict__ agg, const float* __restrict__ hw,
                                const float* __restrict__ deg, const float* __restrict__ bias,
                                const bf16_t* __restrict__ ident, int ldi,
                                bf16_t* __restrict__ out, int ldo, int N, int H) {
  long long t = (long long)blockIdx.x * blockDim.x + threadIdx.x;
  long long total = (long long)N * H;
  if (t >= total) return;
  long long i = t / H;
  int c = (int)(t % H);
  float val = agg[i * H + c] + hw[i * H + c] / deg[i] + bias[c];
  val = lrelu_f(val);
  val += (float)ident[i * ldi + c];
  out[i * ldo + c] = (bf16_t)val;
}

// ---------------------------------------------------------------------------
// Pooling (mean + max per graph) and final projection
// ---------------------------------------------------------------------------
__device__ __forceinline__ void atomicMaxF(float* addr, float v) {
  if (v >= 0.f) atomicMax((int*)addr, __float_as_int(v));
  else          atomicMin((unsigned int*)addr, __float_as_uint(v));
}

__global__ void pool_init_kernel(float* __restrict__ sums, float* __restrict__ maxs,
                                 float* __restrict__ cnt, int G, int F) {
  int t = blockIdx.x * blockDim.x + threadIdx.x;
  int total = G * F;
  if (t < total) { sums[t] = 0.f; maxs[t] = -INFINITY; }
  if (t < G) cnt[t] = 0.f;
}

__global__ void pool_count_kernel(const long long* __restrict__ batch, float* __restrict__ cnt,
                                  int N) {
  int i = blockIdx.x * blockDim.x + threadIdx.x;
  if (i < N) atomicAdd(&cnt[(int)batch[i]], 1.0f);
}

__global__ void pool_acc_kernel(const bf16_t* __restrict__ h, const long long* __restrict__ batch,
                                float* __restrict__ sums, float* __restrict__ maxs, int N, int F) {
  long long t = (long long)blockIdx.x * blockDim.x + threadIdx.x;
  long long total = (long long)N * F;
  if (t >= total) return;
  long long i = t / F;
  int c = (int)(t % F);
  int g = (int)batch[i];
  float v = (float)h[i * F + c];
  atomicAdd(&sums[g * F + c], v);
  atomicMaxF(&maxs[g * F + c], v);
}

__global__ void pool_final_kernel(const float* __restrict__ sums, const float* __restrict__ maxs,
                                  const float* __restrict__ cnt, const float* __restrict__ wpost,
                                  float* __restrict__ out, int F) {
  int g = blockIdx.x;
  int c = threadIdx.x;  // F threads
  float pooled = sums[g * F + c] / cnt[g] + maxs[g * F + c];
  float p = pooled * wpost[c];
  __shared__ float red[128];
  red[c] = p;
  __syncthreads();
  for (int s = F >> 1; s > 0; s >>= 1) {
    if (c < s) red[c] += red[c + s];
    __syncthreads();
  }
  if (c == 0) out[g] = red[0];
}

// ---------------------------------------------------------------------------
// Orchestration
// ---------------------------------------------------------------------------
extern "C" void kernel_launch(void* const* d_in, const int* in_sizes, int n_in,
                              void* d_out, int out_size, void* d_ws, size_t ws_size,
                              hipStream_t stream) {
  const int F_IN = 140, OUT = 128, G = 64;
  const long long N = in_sizes[0] / F_IN;
  const long long E = in_sizes[1] / 2;

  const float*     x      = (const float*)d_in[0];
  const long long* eidx   = (const long long*)d_in[1];
  const long long* src    = eidx;
  const long long* dst    = eidx + E;
  const long long* batch  = (const long long*)d_in[2];
  const float* w_pre1 = (const float*)d_in[4];
  const float* b_pre1 = (const float*)d_in[5];
  const float* w_pre2 = (const float*)d_in[6];
  const float* b_pre2 = (const float*)d_in[7];
  const float* w_res1 = (const float*)d_in[8];
  const float* w_res3 = (const float*)d_in[9];
  const float* w1     = (const float*)d_in[10];
  const float* b1     = (const float*)d_in[11];
  const float* w2     = (const float*)d_in[12];
  const float* b2     = (const float*)d_in[13];
  const float* w3     = (const float*)d_in[14];
  const float* b3     = (const float*)d_in[15];
  const float* w_post = (const float*)d_in[16];
  float* outp = (float*)d_out;

  // ---- workspace carve-up ----
  char* ws = (char*)d_ws;
  size_t off = 0;
  auto take = [&](size_t bytes) -> void* {
    void* p = ws + off;
    off = (off + bytes + 255) & ~(size_t)255;
    return p;
  };
  bf16_t* XB   = (bf16_t*)take((size_t)N * 160 * 2);  // padded x
  bf16_t* T1   = (bf16_t*)take((size_t)N * 256 * 2);  // prenet mid / h1 / h3
  bf16_t* HB0  = (bf16_t*)take((size_t)N * 192 * 2);  // h0 / h2
  bf16_t* HB1  = (bf16_t*)take((size_t)N * 192 * 2);  // ident1 / ident3
  float*  HW32 = (float*) take((size_t)N * 192 * 4);  // conv linear out
  float*  AGG  = (float*) take((size_t)N * 192 * 4);  // aggregation accumulator
  float*  DEG  = (float*) take((size_t)N * 4);
  bf16_t* Wp_pre1 = (bf16_t*)take((size_t)160 * 256 * 2);
  bf16_t* Wp_pre2 = (bf16_t*)take((size_t)256 * 128 * 2);
  bf16_t* Wp_res1 = (bf16_t*)take((size_t)128 * 192 * 2);
  bf16_t* Wp_res3 = (bf16_t*)take((size_t)192 * 128 * 2);
  bf16_t* Wp_1    = (bf16_t*)take((size_t)128 * 192 * 2);
  bf16_t* Wp_2    = (bf16_t*)take((size_t)192 * 192 * 2);
  bf16_t* Wp_3    = (bf16_t*)take((size_t)192 * 128 * 2);
  float* PSUM = (float*)take((size_t)G * OUT * 4);
  float* PMAX = (float*)take((size_t)G * OUT * 4);
  float* PCNT = (float*)take((size_t)G * 4);

  const int TB = 256;
  auto blocks = [&](long long n) { return (unsigned)((n + TB - 1) / TB); };

  // ---- pack weights to bf16 WMMA fragment layout ----
  auto pack = [&](const float* W, bf16_t* Wp, int Kin, int Kout, int Kp) {
    int total = Kp * Kout;
    pack_w_kernel<<<blocks(total), TB, 0, stream>>>(W, Wp, Kin, Kout, total);
  };
  pack(w_pre1, Wp_pre1, 140, 256, 160);
  pack(w_pre2, Wp_pre2, 256, 128, 256);
  pack(w_res1, Wp_res1, 128, 192, 128);
  pack(w_res3, Wp_res3, 192, 128, 192);
  pack(w1,     Wp_1,    128, 192, 128);
  pack(w2,     Wp_2,    192, 192, 192);
  pack(w3,     Wp_3,    192, 128, 192);

  // ---- degrees (shared by all three convs) ----
  deg_init_kernel<<<blocks(N), TB, 0, stream>>>(DEG, (int)N);
  deg_acc_kernel<<<blocks(E), TB, 0, stream>>>(dst, DEG, E);

  // ---- prenet ----
  cvt_x_kernel<<<blocks(N * 160), TB, 0, stream>>>(x, XB, (int)N);

  auto gemm = [&](const bf16_t* A, int lda, const bf16_t* Wp, const float* bias,
                  void* C, int ldc, int Kpad, int Kout, int act, int outbf) {
    dim3 grid((unsigned)((N + 127) / 128), (unsigned)(Kout / 32));
    if (act && outbf)
      gemm_wmma_kernel<1, 1><<<grid, 256, 0, stream>>>(A, lda, Wp, bias, C, ldc, (int)N, Kpad, Kout);
    else if (outbf)
      gemm_wmma_kernel<0, 1><<<grid, 256, 0, stream>>>(A, lda, Wp, bias, C, ldc, (int)N, Kpad, Kout);
    else
      gemm_wmma_kernel<0, 0><<<grid, 256, 0, stream>>>(A, lda, Wp, bias, C, ldc, (int)N, Kpad, Kout);
  };

  // t1 = lrelu(x @ w_pre1 + b_pre1)        [N,256] bf16
  gemm(XB, 160, Wp_pre1, b_pre1, T1, 256, 160, 256, 1, 1);
  // h0 = lrelu(t1 @ w_pre2 + b_pre2)       [N,128] bf16
  gemm(T1, 256, Wp_pre2, b_pre2, HB0, 128, 256, 128, 1, 1);

  auto conv = [&](const bf16_t* hin, int ldh, int Kin, const bf16_t* Wp, const float* bias,
                  const bf16_t* ident, int ldi, bf16_t* hout, int ldo, int Hc) {
    gemm(hin, ldh, Wp, nullptr, HW32, Hc, Kin, Hc, 0, 0);
    (void)hipMemsetAsync(AGG, 0, (size_t)N * Hc * 4, stream);
    scatter_kernel<<<blocks(E * (Hc >> 2)), TB, 0, stream>>>(HW32, src, dst, DEG, AGG, E, Hc);
    finalize_kernel<<<blocks(N * Hc), TB, 0, stream>>>(AGG, HW32, DEG, bias, ident, ldi,
                                                       hout, ldo, (int)N, Hc);
  };

  // block 1: ident = h0 @ w_res1 ; h1 = lrelu(conv1(h0)) + ident  -> T1 (ld 192)
  gemm(HB0, 128, Wp_res1, nullptr, HB1, 192, 128, 192, 0, 1);
  conv(HB0, 128, 128, Wp_1, b1, HB1, 192, T1, 192, 192);

  // block 2: identity residual ; h2 = lrelu(conv2(h1)) + h1  -> HB0 (ld 192)
  conv(T1, 192, 192, Wp_2, b2, T1, 192, HB0, 192, 192);

  // block 3: ident = h2 @ w_res3 ; h3 = lrelu(conv3(h2)) + ident -> T1 (ld 128)
  gemm(HB0, 192, Wp_res3, nullptr, HB1, 128, 192, 128, 0, 1);
  conv(HB0, 192, 192, Wp_3, b3, HB1, 128, T1, 128, 128);

  // ---- pooling: mean + max per graph, then dot with w_post ----
  pool_init_kernel<<<blocks(G * OUT), TB, 0, stream>>>(PSUM, PMAX, PCNT, G, OUT);
  pool_count_kernel<<<blocks(N), TB, 0, stream>>>(batch, PCNT, (int)N);
  pool_acc_kernel<<<blocks(N * OUT), TB, 0, stream>>>(T1, batch, PSUM, PMAX, (int)N, OUT);
  pool_final_kernel<<<G, OUT, 0, stream>>>(PSUM, PMAX, PCNT, w_post, outp, OUT);
}
